// UltimateQuantumML_21869973471237
// MI455X (gfx1250) — matrix operand order repaired and checked by reading, not compiled, not run on hardware
//
#include <hip/hip_runtime.h>
#include <hip/hip_bf16.h>
#include <math.h>

// ---------------------------------------------------------------------------
// MI455X (gfx1250, CDNA5) fused implementation, round 2.
//   * all GEMMs via v_wmma_f32_16x16x32_bf16 (wave32)
//   * 32 batch rows (2 M-tiles) per wave -> each B fragment feeds 2 WMMAs,
//     halving L2 weight traffic (the roofline-binding term)
//   * LDS row strides padded (264/72 bf16, 260 f32) to kill ds bank conflicts
//   * weights pre-transposed + bf16-quantized into d_ws; zero-bias staged too
// ---------------------------------------------------------------------------

typedef __bf16 bf16;
typedef __attribute__((ext_vector_type(8)))  bf16  v8bf;
typedef __attribute__((ext_vector_type(16))) bf16  v16bf;
typedef __attribute__((ext_vector_type(8)))  float v8f;

enum { ACT_NONE = 0, ACT_GELU, ACT_TANH, ACT_SILU, ACT_RELU };

constexpr int SW  = 264;   // wide bf16 activation buffer stride (256 + 8 pad)
constexpr int SS  = 260;   // f32 scratch stride (256 + 4 pad)
constexpr int SV  = 200;   // vcat stride (192 + 8 pad)
constexpr int S64 = 72;    // 64-wide bf16 buffer stride (64 + 8 pad)
constexpr int QS  = 32 * S64;  // per-slot elems for qkv/attb

// ------------------------------- param structs ------------------------------
struct Lin  { const bf16* wT; const float* b; };     // wT: [N][K] bf16 in d_ws
struct LN2  { const float* g; const float* b; };

struct EncP {
  Lin amp1, amp2, amp3, ph1, ph2, wv, wo;
  LN2 aln1, aln2, pln;
  const float* convw[3]; const float* convb[3];
  const float* rfL; const float* rpL;               // rf[-1], rp[-1] (64x3)
};
struct VqcP {
  int L;
  Lin l1[10], l2[10];                               // t2 layers: l1=wv, l2=wo
  LN2 ln[10];
  const float* gate;                                // (L,64,6)
  const bf16* entT[10];                             // tanh pre-applied
  Lin nl1[5], nl2[5];
};
struct MeasP { Lin l1, l2; LN2 ln; const float* l3w; const float* l3b; };
struct AllP {
  EncP enc[3];
  VqcP vqc[3];
  LN2  anorm[3];
  Lin  cq, ck, cv, co;                              // cross attention
  MeasP meas[3];
  const float* ew;
  Lin  meta1;
  const float* meta2w; const float* meta2b;
  const float* zb;                                  // 64 zeros (bias for ent)
  const float* x;
  float* out;
};

// --------------------------- prep kernels -----------------------------------
__global__ void k_prep(const float* __restrict__ src, bf16* __restrict__ dst,
                       int K, int N, int doTanh) {
  int i = blockIdx.x * 256 + threadIdx.x;
  if (i >= K * N) return;
  int n = i / K, k = i - n * K;                     // dst[n*K + k] = src[k*N + n]
  float v = src[(size_t)k * N + n];
  if (doTanh) v = tanhf(v);
  dst[i] = (bf16)v;
}
__global__ void k_zero(float* p, int n) {
  int i = blockIdx.x * 64 + threadIdx.x;
  if (i < n) p[i] = 0.f;
}

// ------------------------------ device helpers ------------------------------
__device__ __forceinline__ v16bf mk16(v8bf lo, v8bf hi) {
  v16bf r;
#pragma unroll
  for (int i = 0; i < 8; ++i) { r[i] = lo[i]; r[i + 8] = hi[i]; }
  return r;
}
__device__ __forceinline__ v16bf load_a(const bf16* arow, int kt, int h) {
  const bf16* pa = arow + kt * 32 + h * 8;          // K = h*8..+7 , 16+h*8..+7
  return mk16(*(const v8bf*)pa, *(const v8bf*)(pa + 16));
}

// C(32xN) = A(32xK bf16 LDS, stride lda) @ B (via B^T [N][K] bf16, L2-resident)
// + bias. Two 16-row M-tiles share every B fragment. f32 out (stride SS).
__device__ __forceinline__ void wave_gemm(const bf16* A, int lda,
                                          const bf16* Bt, int K,
                                          const float* bias,
                                          float* out, int N) {
  const int l = threadIdx.x;
  const int m = l & 15, h = l >> 4;
  const bf16* arow0 = A + m * lda;                  // rows 0..15
  const bf16* arow1 = A + (m + 16) * lda;           // rows 16..31
  for (int nt = 0; nt < (N >> 4); ++nt) {
    v8f c0 = {0.f,0.f,0.f,0.f,0.f,0.f,0.f,0.f};
    v8f c1 = {0.f,0.f,0.f,0.f,0.f,0.f,0.f,0.f};
    const bf16* brow = Bt + (size_t)(nt * 16 + m) * K + h * 8;  // col n=m, K = h*8..+7 , 16+h*8..+7
    for (int kt = 0; kt < (K >> 5); ++kt) {
      const bf16* pb = brow + kt * 32;
      v16bf b = mk16(*(const v8bf*)pb, *(const v8bf*)(pb + 16));
      c0 = __builtin_amdgcn_wmma_f32_16x16x32_bf16(false, load_a(arow0, kt, h),
                                                   false, b, (short)0, c0, false, false);
      c1 = __builtin_amdgcn_wmma_f32_16x16x32_bf16(false, load_a(arow1, kt, h),
                                                   false, b, (short)0, c1, false, false);
    }
    float bb = bias[nt * 16 + m];
#pragma unroll
    for (int r = 0; r < 8; ++r) {                   // C: VGPR r -> row r+8h, col lane&15
      out[(r + 8 * h) * SS + nt * 16 + m]      = c0[r] + bb;
      out[(16 + r + 8 * h) * SS + nt * 16 + m] = c1[r] + bb;
    }
  }
}

__device__ __forceinline__ float act_f(float v, int a) {
  switch (a) {
    case ACT_GELU: return 0.5f * v * (1.0f + erff(v * 0.70710678118f));
    case ACT_TANH: return tanhf(v);
    case ACT_SILU: return v / (1.0f + __expf(-v));
    case ACT_RELU: return fmaxf(v, 0.0f);
  }
  return v;
}

// optional LayerNorm (eps 1e-5) + activation + bf16 store (32 rows x N)
__device__ __forceinline__ void epi(const float* scr, int N,
                                    const float* g, const float* b, float* rs,
                                    int a, bf16* dst, int ldd) {
  const int l = threadIdx.x;
  if (g) {                                          // lane l owns row l, staggered cols
    float s = 0.f, s2 = 0.f;
    int c = l % N;
    for (int i = 0; i < N; ++i) {
      float v = scr[l * SS + c];
      s += v; s2 += v * v;
      if (++c == N) c = 0;
    }
    float mm = s / N;
    rs[l]      = mm;
    rs[32 + l] = rsqrtf(fmaxf(s2 / N - mm * mm, 0.f) + 1e-5f);
    __syncthreads();
  }
  for (int t = l; t < 32 * N; t += 32) {
    int row = t / N, c = t - row * N;
    float v = scr[row * SS + c];
    if (g) v = (v - rs[row]) * rs[32 + row] * g[c] + b[c];
    dst[row * ldd + c] = (bf16)act_f(v, a);
  }
  __syncthreads();
}

// ------------------------------- main kernel --------------------------------
__global__ void __launch_bounds__(32)
k_main(AllP P) {
  // manually-managed LDS arena (phase overlays), ~121 KB
  __shared__ __align__(16) unsigned char sh[124160];
  bf16*  xbf   = (bf16*)(sh + 0);          // 32 x SW           (16896 B)
  bf16*  bufA  = (bf16*)(sh + 16896);      // 32 x SW
  bf16*  bufB  = (bf16*)(sh + 33792);      // 32 x SW
  float* convf = (float*)(sh + 50688);     // 32 x 64 f32       (8192 B)
  bf16*  resb  = (bf16*)(sh + 58880);      // 32 x S64          (4608 B)
  bf16*  qkv   = (bf16*)(sh + 0);          // 9 x QS  (overlay; cross phase only)
  float* scr   = (float*)(sh + 63488);     // 32 x SS f32       (33280 B)
  bf16*  vcat  = (bf16*)(sh + 96768);      // 32 x SV           (12800 B)
  bf16*  attb  = (bf16*)(sh + 109568);     // 3 x QS            (13824 B)
  float* rs    = (float*)(sh + 123392);    // 64 row stats
  float* wbar  = (float*)(sh + 123648);    // [3][8] FIR taps
  float* bbar  = (float*)(sh + 123744);    // [3]
  float* ptb   = (float*)(sh + 123760);    // 32 x 3 preds

  const int l = threadIdx.x;
  const size_t row0 = (size_t)blockIdx.x * 32;

  // ---- load x tile, quantize to bf16 ----
  for (int t = l; t < 32 * 64; t += 32) {           // 64 float4 per row
    int row = t >> 6, c4 = t & 63;
    const float4 v = ((const float4*)(P.x + (row0 + row) * 256))[c4];
    bf16* d = &xbf[row * SW + c4 * 4];
    d[0] = (bf16)v.x; d[1] = (bf16)v.y; d[2] = (bf16)v.z; d[3] = (bf16)v.w;
  }
  __syncthreads();

  const int ks3[3] = {3, 5, 7};

  // =========================== ensemble loop ===========================
  for (int i = 0; i < 3; ++i) {
    const EncP& E = P.enc[i];

    // ---- conv branch: channel-mean folded into per-kernel FIR taps ----
    if (l < 21) {
      int kk = l / 7, t = l - kk * 7, k = ks3[kk];
      float s = 0.f;
      if (t < k) { const float* w = E.convw[kk]; for (int c = 0; c < 16; ++c) s += w[c * k + t]; }
      wbar[kk * 8 + t] = s * (1.f / 16.f);
    } else if (l < 24) {
      int kk = l - 21; const float* b = E.convb[kk];
      float s = 0.f; for (int c = 0; c < 16; ++c) s += b[c];
      bbar[kk] = s * (1.f / 16.f);
    }
    __syncthreads();
    for (int t = l; t < 32 * 64; t += 32) {
      int row = t >> 6, e = t & 63;
      float acc = 0.f;
      for (int kk = 0; kk < 3; ++kk) {
        int k = ks3[kk], pad = k >> 1; float vk = 0.f;
        for (int j = 0; j < 4; ++j) {
          int p = e * 4 + j; float f = 0.f;
          for (int tt = 0; tt < k; ++tt) {
            int xi = p + tt - pad;
            if (xi >= 0 && xi < 256) f += wbar[kk * 8 + tt] * (float)xbf[row * SW + xi];
          }
          vk += f;
        }
        acc += bbar[kk] + 0.25f * vk;
      }
      convf[t] = acc * (1.f / 3.f);
    }
    __syncthreads();

    // ---- amplitude / phase MLPs ----
    wave_gemm(xbf, SW, E.amp1.wT, 256, E.amp1.b, scr, 256);
    epi(scr, 256, E.aln1.g, E.aln1.b, rs, ACT_GELU, bufA, SW);
    wave_gemm(xbf, SW, E.ph1.wT, 256, E.ph1.b, scr, 128);
    epi(scr, 128, E.pln.g, E.pln.b, rs, ACT_SILU, bufB, SW);
    wave_gemm(bufA, SW, E.amp2.wT, 256, E.amp2.b, scr, 128);
    epi(scr, 128, E.aln2.g, E.aln2.b, rs, ACT_TANH, bufA, SW);
    wave_gemm(bufB, SW, E.ph2.wT, 128, E.ph2.b, scr, 64);
    epi(scr, 64, nullptr, nullptr, nullptr, ACT_TANH, bufB, SW);
    wave_gemm(bufA, SW, E.amp3.wT, 128, E.amp3.b, scr, 64);  // a3 f32 -> scr[:,0:64]
    __syncthreads();

    // ---- qs = (sin(a*rf0+rp0)+cos(ph*rf1+rp1)+tanh(conv*rf2+rp2))/3 ----
    for (int t = l; t < 32 * 64; t += 32) {
      int row = t >> 6, e = t & 63;
      float a3 = scr[row * SS + e];
      float ph = (float)bufB[row * SW + e];
      float cv = convf[t];
      const float* rf = E.rfL + e * 3; const float* rp = E.rpL + e * 3;
      float q = (sinf(a3 * rf[0] + rp[0]) + cosf(ph * rf[1] + rp[1]) +
                 tanhf(cv * rf[2] + rp[2])) * (1.f / 3.f);
      bufA[row * SW + e] = (bf16)q;
    }
    __syncthreads();

    // ---- encoder MHA with L=1  ->  (qs @ wv + bv) @ wo + bo ----
    wave_gemm(bufA, SW, E.wv.wT, 64, E.wv.b, scr, 64);
    epi(scr, 64, nullptr, nullptr, nullptr, ACT_NONE, bufB, SW);
    wave_gemm(bufB, SW, E.wo.wT, 64, E.wo.b, scr, 64);
    epi(scr, 64, nullptr, nullptr, nullptr, ACT_NONE, bufA, SW);   // s0

    // =========================== VQC layers ===========================
    const VqcP& V = P.vqc[i];
    for (int lay = 0; lay < V.L; ++lay) {
      int t3 = lay % 3;
      for (int t = l; t < 32 * 64; t += 32)                       // residual save
        resb[(t >> 6) * S64 + (t & 63)] = bufA[(t >> 6) * SW + (t & 63)];
      __syncthreads();

      if (t3 == 0) {
        wave_gemm(bufA, SW, V.l1[lay].wT, 64, V.l1[lay].b, scr, 192);
        epi(scr, 192, V.ln[lay].g, V.ln[lay].b, rs, ACT_GELU, bufB, SW);
        wave_gemm(bufB, SW, V.l2[lay].wT, 192, V.l2[lay].b, scr + 192, 64);
      } else if (t3 == 1) {
        wave_gemm(bufA, SW, V.l1[lay].wT, 64, V.l1[lay].b, scr, 128);
        epi(scr, 128, nullptr, nullptr, nullptr, ACT_SILU, bufB, SW);
        wave_gemm(bufB, SW, V.l2[lay].wT, 128, V.l2[lay].b, scr + 192, 64);
      } else {                                                    // L=1 MHA
        wave_gemm(bufA, SW, V.l1[lay].wT, 64, V.l1[lay].b, scr, 64);
        epi(scr, 64, nullptr, nullptr, nullptr, ACT_NONE, bufB, SW);
        wave_gemm(bufB, SW, V.l2[lay].wT, 64, V.l2[lay].b, scr + 192, 64);
      }
      __syncthreads();

      // s' (f32 at scr col 192) -> bf16 for ent GEMM;  ent = s' @ tanh(W_ent)
      epi(scr + 192, 64, nullptr, nullptr, nullptr, ACT_NONE, bufB, SW);
      wave_gemm(bufB, SW, V.entT[lay], 64, P.zb, scr, 64);
      __syncthreads();

      // gate + combine: s = (s + 0.3*gate + 0.2*ent)/1.5
      const float* G = V.gate + (size_t)lay * 64 * 6;
      for (int t = l; t < 32 * 64; t += 32) {
        int row = t >> 6, e = t & 63;
        float sp = scr[row * SS + 192 + e];
        float ev = scr[row * SS + e];
        const float* g6 = G + e * 6;
        float gate = ((sinf(g6[0]) + cosf(g6[1]) + tanhf(g6[2])) * sp +
                      0.5f * sinf(g6[3] * sp + g6[4]) + 0.5f * cosf(g6[5] * sp)) * 0.25f;
        float sv = (sp + 0.3f * gate + 0.2f * ev) * (1.0f / 1.5f);
        scr[row * SS + 192 + e] = sv;
        bufA[row * SW + e] = (bf16)sv;
      }
      __syncthreads();

      if (lay & 1) {                                              // nl refinement
        int j = lay >> 1;
        wave_gemm(bufA, SW, V.nl1[j].wT, 64, V.nl1[j].b, scr + 64, 64);
        epi(scr + 64, 64, nullptr, nullptr, nullptr, ACT_TANH, bufB, SW);
        wave_gemm(bufB, SW, V.nl2[j].wT, 64, V.nl2[j].b, scr + 128, 64);
        __syncthreads();
        for (int t = l; t < 32 * 64; t += 32) {
          int row = t >> 6, e = t & 63;
          scr[row * SS + 192 + e] += 0.1f * scr[row * SS + 128 + e];
        }
        __syncthreads();
      }

      // alpha-blend residual, L2-normalize row, tanh -> next s (bf16)
      float alpha = (lay < V.L / 2) ? 0.8f : 0.6f;
      for (int t = l; t < 32 * 64; t += 32) {
        int row = t >> 6, e = t & 63;
        float sv = alpha * scr[row * SS + 192 + e] +
                   (1.f - alpha) * (float)resb[row * S64 + e];
        scr[row * SS + 192 + e] = sv;
      }
      __syncthreads();
      {
        float s2 = 0.f; int c = l & 63;                           // lane l = row l
        for (int i2 = 0; i2 < 64; ++i2) {
          float v = scr[l * SS + 192 + c]; s2 += v * v;
          c = (c + 1) & 63;
        }
        rs[l] = 1.0f / (sqrtf(s2) + 1e-8f);
      }
      __syncthreads();
      for (int t = l; t < 32 * 64; t += 32) {
        int row = t >> 6, e = t & 63;
        bufA[row * SW + e] = (bf16)tanhf(scr[row * SS + 192 + e] * rs[row]);
      }
      __syncthreads();
    }

    // vout_i = LN(s, anorm_i) -> vcat[:, i*64 : i*64+64]
    for (int t = l; t < 32 * 64; t += 32) {
      int row = t >> 6, e = t & 63;
      scr[row * SS + e] = (float)bufA[row * SW + e];
    }
    __syncthreads();
    epi(scr, 64, P.anorm[i].g, P.anorm[i].b, rs, ACT_NONE, vcat + i * 64, SV);
  }

  // ====================== cross attention (L=3, h=4) ======================
  // NOTE: qkv overlays xbf/bufA/bufB region -- all dead here.
  for (int tok = 0; tok < 3; ++tok) {
    wave_gemm(vcat + tok * 64, SV, P.cq.wT, 64, P.cq.b, scr, 64);
    epi(scr, 64, nullptr, nullptr, nullptr, ACT_NONE, qkv + (tok * 3 + 0) * QS, S64);
    wave_gemm(vcat + tok * 64, SV, P.ck.wT, 64, P.ck.b, scr, 64);
    epi(scr, 64, nullptr, nullptr, nullptr, ACT_NONE, qkv + (tok * 3 + 1) * QS, S64);
    wave_gemm(vcat + tok * 64, SV, P.cv.wT, 64, P.cv.b, scr, 64);
    epi(scr, 64, nullptr, nullptr, nullptr, ACT_NONE, qkv + (tok * 3 + 2) * QS, S64);
  }
  // 3x3 scores + softmax + weighted V per (row, head); overwrite q slot with o
  for (int it = 0; it < 4; ++it) {
    int task = l + 32 * it, m = task & 31, hd = task >> 5;        // hd 0..3
    float sc[3][3];
    for (int a = 0; a < 3; ++a)
      for (int b = 0; b < 3; ++b) {
        float s = 0.f;
        for (int d = 0; d < 16; ++d)
          s += (float)qkv[(a * 3 + 0) * QS + m * S64 + hd * 16 + d] *
               (float)qkv[(b * 3 + 1) * QS + m * S64 + hd * 16 + d];
        sc[a][b] = s * 0.25f;                                     // 1/sqrt(16)
      }
    for (int a = 0; a < 3; ++a) {
      float mx = fmaxf(sc[a][0], fmaxf(sc[a][1], sc[a][2]));
      float e0 = __expf(sc[a][0] - mx), e1 = __expf(sc[a][1] - mx), e2 = __expf(sc[a][2] - mx);
      float inv = 1.f / (e0 + e1 + e2);
      float a0 = e0 * inv, a1 = e1 * inv, a2 = e2 * inv;
      for (int d = 0; d < 16; ++d) {
        float o = a0 * (float)qkv[(0 * 3 + 2) * QS + m * S64 + hd * 16 + d] +
                  a1 * (float)qkv[(1 * 3 + 2) * QS + m * S64 + hd * 16 + d] +
                  a2 * (float)qkv[(2 * 3 + 2) * QS + m * S64 + hd * 16 + d];
        qkv[(a * 3 + 0) * QS + m * S64 + hd * 16 + d] = (bf16)o;  // q slot := o
      }
    }
  }
  __syncthreads();
  for (int tok = 0; tok < 3; ++tok) {
    wave_gemm(qkv + (tok * 3 + 0) * QS, S64, P.co.wT, 64, P.co.b, scr, 64);
    epi(scr, 64, nullptr, nullptr, nullptr, ACT_NONE, attb + tok * QS, S64);
  }

  // ========================= measurement heads =========================
  // (qkv region dead again -> bufA/bufB reuse is safe)
  for (int i = 0; i < 3; ++i) {
    const MeasP& M = P.meas[i];
    wave_gemm(attb + i * QS, S64, M.l1.wT, 64, M.l1.b, scr, 128);
    epi(scr, 128, M.ln.g, M.ln.b, rs, ACT_GELU, bufA, SW);
    wave_gemm(bufA, SW, M.l2.wT, 128, M.l2.b, scr, 32);
    epi(scr, 32, nullptr, nullptr, nullptr, ACT_SILU, bufB, SW);
    {
      float a = M.l3b[0];
      for (int j = 0; j < 32; ++j) a += (float)bufB[l * SW + j] * M.l3w[j];
      ptb[l * 3 + i] = a;                                         // lane l = row l
    }
    __syncthreads();
  }

  // ============================ meta gating ============================
  wave_gemm(vcat, SV, P.meta1.wT, 192, P.meta1.b, scr, 64);
  epi(scr, 64, nullptr, nullptr, nullptr, ACT_RELU, bufA, SW);
  {
    float z[3];
    for (int o = 0; o < 3; ++o) {
      float a = P.meta2b[o];
      for (int j = 0; j < 64; ++j) a += (float)bufA[l * SW + j] * P.meta2w[j * 3 + o];
      z[o] = a;
    }
    float mx = fmaxf(z[0], fmaxf(z[1], z[2]));
    float e0 = __expf(z[0] - mx), e1 = __expf(z[1] - mx), e2 = __expf(z[2] - mx);
    float inv = 1.f / (e0 + e1 + e2);
    float fin = ptb[l * 3 + 0] * e0 * inv + ptb[l * 3 + 1] * e1 * inv + ptb[l * 3 + 2] * e2 * inv;
    float ewd = ptb[l * 3 + 0] * P.ew[0] + ptb[l * 3 + 1] * P.ew[1] + ptb[l * 3 + 2] * P.ew[2];
    P.out[row0 + l] = 0.5f * (fin + ewd);
  }
}

// ------------------------------- host side ---------------------------------
static const bf16* stageW(const float* w, int K, int N, int doTanh,
                          bf16* wsb, size_t& off, hipStream_t s) {
  bf16* dst = wsb + off;
  size_t n = (size_t)K * N;
  off += (n + 15) & ~(size_t)15;
  k_prep<<<dim3((unsigned)((n + 255) / 256)), dim3(256), 0, s>>>(w, dst, K, N, doTanh);
  return dst;
}

extern "C" void kernel_launch(void* const* d_in, const int* in_sizes, int n_in,
                              void* d_out, int out_size, void* d_ws, size_t ws_size,
                              hipStream_t stream) {
  (void)out_size; (void)ws_size;
  // Leaves arrive flat in setup_inputs() construction order; x detected by size.
  int idx = 0;
  const bool x_first = (in_sizes[0] == 32768 * 256);
  const float* xptr = x_first ? (const float*)d_in[0] : (const float*)d_in[n_in - 1];
  if (x_first) idx = 1;
  auto F = [&]() -> const float* {
    return (const float*)d_in[(idx < n_in) ? idx++ : 0];
  };
  bf16* wsb = (bf16*)d_ws;
  size_t off = 0;

  // zero bias vector (for the bias-free ent GEMM) -> branchless epilogue
  float* zb = (float*)wsb;
  off += 256;                                   // 128 floats reserved
  k_zero<<<dim3(1), dim3(64), 0, stream>>>(zb, 64);

  auto ST = [&](int K, int N, int th) -> const bf16* {
    const float* w = F();
    return stageW(w, K, N, th, wsb, off, stream);
  };

  AllP P{};
  for (int i = 0; i < 3; ++i) {
    EncP& E = P.enc[i];
    E.amp1.wT = ST(256, 256, 0); E.amp1.b = F(); E.aln1.g = F(); E.aln1.b = F();
    E.amp2.wT = ST(256, 128, 0); E.amp2.b = F(); E.aln2.g = F(); E.aln2.b = F();
    E.amp3.wT = ST(128, 64, 0);  E.amp3.b = F();
    E.ph1.wT  = ST(256, 128, 0); E.ph1.b  = F(); E.pln.g  = F(); E.pln.b  = F();
    E.ph2.wT  = ST(128, 64, 0);  E.ph2.b  = F();
    const float* rf = F(); const float* rp = F();                 // (5+i, 64, 3)
    E.rfL = rf + (size_t)(4 + i) * 192;
    E.rpL = rp + (size_t)(4 + i) * 192;
    for (int kk = 0; kk < 3; ++kk) { E.convw[kk] = F(); E.convb[kk] = F(); }
    F(); F(); F(); F();                                           // wq,bq,wk,bk unused
    E.wv.wT = ST(64, 64, 0); E.wv.b = F();
    E.wo.wT = ST(64, 64, 0); E.wo.b = F();
  }
  for (int i = 0; i < 3; ++i) {
    VqcP& V = P.vqc[i]; V.L = 8 + i;
    for (int lay = 0; lay < V.L; ++lay) {
      int t = lay % 3;
      if (t == 0) {
        V.l1[lay].wT = ST(64, 192, 0); V.l1[lay].b = F();
        V.ln[lay].g = F(); V.ln[lay].b = F();
        V.l2[lay].wT = ST(192, 64, 0); V.l2[lay].b = F();
      } else if (t == 1) {
        V.l1[lay].wT = ST(64, 128, 0); V.l1[lay].b = F();
        V.l2[lay].wT = ST(128, 64, 0); V.l2[lay].b = F();
      } else {
        F(); F(); F(); F();                                       // wq,bq,wk,bk unused
        V.l1[lay].wT = ST(64, 64, 0); V.l1[lay].b = F();          // wv,bv
        V.l2[lay].wT = ST(64, 64, 0); V.l2[lay].b = F();          // wo,bo
      }
    }
    V.gate = F();
    for (int lay = 0; lay < V.L; ++lay) V.entT[lay] = ST(64, 64, 1); // tanh fused
    for (int j = 0; j < V.L / 2; ++j) {
      V.nl1[j].wT = ST(64, 64, 0); V.nl1[j].b = F();
      V.nl2[j].wT = ST(64, 64, 0); V.nl2[j].b = F();
    }
  }
  for (int i = 0; i < 3; ++i) { P.anorm[i].g = F(); P.anorm[i].b = F(); }
  P.cq.wT = ST(64, 64, 0); P.cq.b = F();
  P.ck.wT = ST(64, 64, 0); P.ck.b = F();
  P.cv.wT = ST(64, 64, 0); P.cv.b = F();
  P.co.wT = ST(64, 64, 0); P.co.b = F();
  for (int i = 0; i < 3; ++i) {
    MeasP& M = P.meas[i];
    M.l1.wT = ST(64, 128, 0); M.l1.b = F(); M.ln.g = F(); M.ln.b = F();
    M.l2.wT = ST(128, 32, 0); M.l2.b = F();
    M.l3w = F(); M.l3b = F();
  }
  P.ew = F();
  P.meta1.wT = ST(192, 64, 0); P.meta1.b = F();
  P.meta2w = F(); P.meta2b = F();
  P.zb = zb;
  P.x = xptr;
  P.out = (float*)d_out;

  const int xN = x_first ? in_sizes[0] : in_sizes[n_in - 1];
  const int nrows = xN / 256;                                     // B = 32768
  k_main<<<dim3(nrows / 32), dim3(32), 0, stream>>>(P);
}
